// VisualAttention_14216341749871
// MI455X (gfx1250) — matrix operand order repaired and checked by reading, not compiled, not run on hardware
//
#include <hip/hip_runtime.h>
#include <math.h>

typedef __attribute__((ext_vector_type(16))) __bf16        v16bf;
typedef __attribute__((ext_vector_type(8)))  float         v8f;
typedef __attribute__((ext_vector_type(4)))  unsigned int  u32x4;

// ---------------------------------------------------------------------------
// A-fragment K index for 16-bit A 16x32 (ISA 7.12.2):
//   VGPR r (pair 2r,2r+1): r<4 -> K = lg*8 + 2r ; r>=4 -> K = 16 + lg*8 + 2(r-4)
// ---------------------------------------------------------------------------
__device__ __forceinline__ int a_k0(int r, int lg) {
    return (r < 4) ? (lg * 8 + 2 * r) : (16 + lg * 8 + 2 * (r - 4));
}

__device__ __forceinline__ unsigned int pack_bf16(float a, float b) {
    union { __bf16 h[2]; unsigned int u; } x;
    x.h[0] = (__bf16)a;
    x.h[1] = (__bf16)b;
    return x.u;
}

__device__ __forceinline__ v16bf frag_from(u32x4 lo, u32x4 hi) {
    union { u32x4 u[2]; v16bf f; } x;
    x.u[0] = lo;
    x.u[1] = hi;
    return x.f;
}

// ===========================================================================
// Kernel 1: projection GEMM  Y[o,n] = sum_c W[o,c] * X[c,n] + bias[o]
//   W: [O, C] f32 row-major, X: [B, C, N] f32, Y: [B, O, N] bf16
//   One wave computes one 16x16 tile of Y, K-loop over C in steps of 32.
// ===========================================================================
__global__ void proj_gemm(const float* __restrict__ W, const float* __restrict__ bias,
                          const float* __restrict__ X, __bf16* __restrict__ Y,
                          int O, int N, int C) {
    const int lane = threadIdx.x & 31;
    const int lg = lane >> 4, lm = lane & 15;
    const int ntiles = N / 16;
    const int mo = blockIdx.x / ntiles;   // o-tile
    const int nt = blockIdx.x % ntiles;   // n-tile
    const int b  = blockIdx.y;
    const float* Xb = X + (size_t)b * C * N;
    __bf16* Yb = Y + (size_t)b * O * N;
    const int obase = mo * 16, nbase = nt * 16;

    v8f acc = {};
    for (int kk = 0; kk < C; kk += 32) {
        v16bf A, Bm;
        // A[m][k] = W[(obase+lm)*C + kk + k]  (contiguous K pairs)
        #pragma unroll
        for (int r = 0; r < 8; ++r) {
            const float* p = W + (size_t)(obase + lm) * C + kk + a_k0(r, lg);
            A[2 * r]     = (__bf16)p[0];
            A[2 * r + 1] = (__bf16)p[1];
        }
        // B[k][j] = X[(kk+k)*N + nbase + lm]  (k strided by N)
        #pragma unroll
        for (int r = 0; r < 8; ++r) {
            int k0 = a_k0(r, lg);
            Bm[2 * r]     = (__bf16)Xb[(size_t)(kk + k0) * N + nbase + lm];
            Bm[2 * r + 1] = (__bf16)Xb[(size_t)(kk + k0 + 1) * N + nbase + lm];
        }
        acc = __builtin_amdgcn_wmma_f32_16x16x32_bf16(false, A, false, Bm,
                                                      (short)0, acc, false, false);
    }
    // D[m][n]: m = r + lg*8, n = lm ; fuse bias, convert to bf16
    #pragma unroll
    for (int r = 0; r < 8; ++r) {
        int m = obase + r + lg * 8;
        float val = acc[r] + bias[m];
        Yb[(size_t)m * N + nbase + lm] = (__bf16)val;
    }
}

// ===========================================================================
// Kernel 2: energy tile  E[m,n] = SCALE * sum_d q[d,m] * k[d,n]
//   Q,K stored d-major [B, 32, N] bf16 -> K-dim = 32 = single WMMA per tile.
//   One wave: one 16-row m-tile x NQT n-tiles, A-fragment loaded once.
//   Writes pre-softmax logits into the attention output region (f32).
// ===========================================================================
#define NQT 8
__global__ void energy_kernel(const __bf16* __restrict__ Q, const __bf16* __restrict__ K,
                              float* __restrict__ E, int N, float scale) {
    const int lane = threadIdx.x & 31;
    const int lg = lane >> 4, lm = lane & 15;
    const int mt = blockIdx.x;
    const int ng = blockIdx.y;
    const int b  = blockIdx.z;
    const __bf16* Qb = Q + (size_t)b * 32 * N;
    const __bf16* Kb = K + (size_t)b * 32 * N;
    float* Eb = E + (size_t)b * N * N;
    const int mbase = mt * 16;

    v16bf A;
    #pragma unroll
    for (int r = 0; r < 8; ++r) {
        int k0 = a_k0(r, lg);
        A[2 * r]     = Qb[(size_t)k0 * N + mbase + lm];
        A[2 * r + 1] = Qb[(size_t)(k0 + 1) * N + mbase + lm];
    }
    for (int t = 0; t < NQT; ++t) {
        const int nbase = (ng * NQT + t) * 16;
        v16bf Bm;
        #pragma unroll
        for (int r = 0; r < 8; ++r) {
            int k0 = a_k0(r, lg);
            Bm[2 * r]     = Kb[(size_t)k0 * N + nbase + lm];
            Bm[2 * r + 1] = Kb[(size_t)(k0 + 1) * N + nbase + lm];
        }
        v8f acc = {};
        acc = __builtin_amdgcn_wmma_f32_16x16x32_bf16(false, A, false, Bm,
                                                      (short)0, acc, false, false);
        #pragma unroll
        for (int r = 0; r < 8; ++r) {
            int m = mbase + r + lg * 8;
            Eb[(size_t)m * N + nbase + lm] = acc[r] * scale;
        }
    }
}

// ===========================================================================
// Kernel 3: in-place row softmax over N=4096 (one 256-thread block per row)
// ===========================================================================
__global__ void softmax_rows(float* __restrict__ A, int N) {
    __shared__ float red[256];
    float* p = A + (size_t)blockIdx.x * N;
    const int tid = threadIdx.x;

    float mx = -INFINITY;
    for (int i = tid; i < N; i += 256) mx = fmaxf(mx, p[i]);
    red[tid] = mx; __syncthreads();
    for (int s = 128; s > 0; s >>= 1) {
        if (tid < s) red[tid] = fmaxf(red[tid], red[tid + s]);
        __syncthreads();
    }
    mx = red[0]; __syncthreads();

    float sum = 0.f;
    for (int i = tid; i < N; i += 256) {
        float e = __expf(p[i] - mx);
        p[i] = e;
        sum += e;
    }
    red[tid] = sum; __syncthreads();
    for (int s = 128; s > 0; s >>= 1) {
        if (tid < s) red[tid] += red[tid + s];
        __syncthreads();
    }
    float inv = 1.0f / red[0];
    for (int i = tid; i < N; i += 256) p[i] *= inv;
}

// ===========================================================================
// Kernel 4: texture GEMM  Tex[c,m] = sum_n v[c,n] * attn[m,n], K = N = 4096
//   Block = 256 threads (8 waves) computing a 128c x 64m tile.
//   Per 64-wide K chunk the block stages the 64m x 64k attention sub-tile:
//   fp32 global_load_b128 -> v_cvt_pk_bf16_f32 ONCE -> bf16 in LDS
//   (row stride 72 bf16 = 144 B: 16B-aligned b128, conflict-free across lanes).
//   Hot loop per wmma group: 2x ds_load_b128 (B-frag, pure 32B copy in exact
//   fragment element order) + 2x global_load_b128 (A-frag v, two aligned
//   8-element K runs per lane). Attention read 2x from HBM total (vs 16x
//   untiled), conversions amortized 8x. Fuses residual add: Out = Tex + SK.
// ===========================================================================
#define TK 64                 // K chunk staged in LDS
#define TM 64                 // m columns per block
#define LDS_STRIDE_U32 36     // (TK + 8 pad) bf16 = 72 bf16 = 36 dwords = 144 B
#define MT 4                  // m-tiles (16 cols) per wave

__global__ __launch_bounds__(256) void texture_kernel(
        const __bf16* __restrict__ V, const float* __restrict__ Attn,
        const float* __restrict__ SK, float* __restrict__ Tex,
        float* __restrict__ Out, int N, int C) {
    __shared__ unsigned int atile[TM * LDS_STRIDE_U32];   // 9.2 KB bf16 tile

    const int tid  = threadIdx.x;
    const int wave = tid >> 5;
    const int lane = tid & 31;
    const int lg = lane >> 4, lm = lane & 15;
    const int mg = blockIdx.x;               // m-group (TM cols)
    const int cg = blockIdx.y;               // c-group (128 rows)
    const int b  = blockIdx.z;
    const __bf16* Vb = V + (size_t)b * C * N;
    const float*  Ab = Attn + (size_t)b * N * N;
    const int cbase  = cg * 128 + wave * 16;
    const int mblock = mg * TM;

    // Staging: thread -> (row, 16-float segment); 256 threads cover 64 rows x
    // 4 segments. Each thread: 4x global b128 (f32) -> 8 packed bf16 dwords ->
    // 2x LDS b128 stores.
    const int srow = tid >> 2;
    const int sseg = (tid & 3) * 16;          // float index within row

    v8f acc[MT] = {};
    for (int kk = 0; kk < N; kk += TK) {
        __syncthreads();   // previous chunk's readers done before overwrite
        {
            const float4* src =
                (const float4*)(Ab + (size_t)(mblock + srow) * N + kk + sseg);
            float4 f0 = src[0], f1 = src[1], f2 = src[2], f3 = src[3];
            u32x4 lo = { pack_bf16(f0.x, f0.y), pack_bf16(f0.z, f0.w),
                         pack_bf16(f1.x, f1.y), pack_bf16(f1.z, f1.w) };
            u32x4 hi = { pack_bf16(f2.x, f2.y), pack_bf16(f2.z, f2.w),
                         pack_bf16(f3.x, f3.y), pack_bf16(f3.z, f3.w) };
            u32x4* dst = (u32x4*)&atile[srow * LDS_STRIDE_U32 + (sseg >> 1)];
            dst[0] = lo;
            dst[1] = hi;
            if (kk + TK < N) {  // prefetch next attention chunk toward L2
                __builtin_prefetch(Ab + (size_t)(mblock + srow) * N + kk + TK + sseg, 0, 1);
            }
        }
        __syncthreads();

        #pragma unroll
        for (int k2 = 0; k2 < TK; k2 += 32) {
            // A-fragment: v (bf16) — per lane two contiguous 16B K runs
            const __bf16* vrow = Vb + (size_t)(cbase + lm) * N + kk + k2;
            u32x4 alo = *(const u32x4*)(vrow + lg * 16);
            u32x4 ahi = *(const u32x4*)(vrow + lg * 16 + 8);
            v16bf Afrag = frag_from(alo, ahi);

            #pragma unroll
            for (int t = 0; t < MT; ++t) {
                // B-fragment: 16 consecutive bf16 of LDS row (t*16+lm) at
                // column k2 + lg*16 — exact fragment element order.
                const u32x4* p = (const u32x4*)
                    &atile[(t * 16 + lm) * LDS_STRIDE_U32 + ((k2 + lg * 16) >> 1)];
                v16bf Bfrag = frag_from(p[0], p[1]);
                acc[t] = __builtin_amdgcn_wmma_f32_16x16x32_bf16(
                    false, Afrag, false, Bfrag, (short)0, acc[t], false, false);
            }
        }
    }

    const float* SKb = SK + (size_t)b * C * N;
    float* Txb = Tex + (size_t)b * C * N;
    float* Ob  = Out + (size_t)b * C * N;
    #pragma unroll
    for (int t = 0; t < MT; ++t) {
        #pragma unroll
        for (int r = 0; r < 8; ++r) {
            int c = cbase + r + lg * 8;
            int m = mblock + t * 16 + lm;
            float tv = acc[t][r];
            Txb[(size_t)c * N + m] = tv;
            Ob[(size_t)c * N + m]  = tv + SKb[(size_t)c * N + m];
        }
    }
}

// ===========================================================================
extern "C" void kernel_launch(void* const* d_in, const int* in_sizes, int n_in,
                              void* d_out, int out_size, void* d_ws, size_t ws_size,
                              hipStream_t stream) {
    const float* sk = (const float*)d_in[0];
    const float* ex = (const float*)d_in[1];
    const float* Wq = (const float*)d_in[2];
    const float* bq = (const float*)d_in[3];
    const float* Wk = (const float*)d_in[4];
    const float* bk = (const float*)d_in[5];
    const float* Wv = (const float*)d_in[6];
    const float* bv = (const float*)d_in[7];

    const int B = 8, C = 256, N = 4096, CQK = 32;
    const float SCALE = 0.1767766952966369f;  // 1/sqrt(32)

    // Output layout: out [B,C,N] | texture [B,C,N] | attention [B,N,N]
    float* out  = (float*)d_out;
    float* tex  = out + (size_t)B * C * N;
    float* attn = tex + (size_t)B * C * N;

    // Workspace: bf16 q [B,32,N] | k [B,32,N] | v [B,256,N]  (~20 MB)
    __bf16* qws = (__bf16*)d_ws;
    __bf16* kws = qws + (size_t)B * CQK * N;
    __bf16* vws = kws + (size_t)B * CQK * N;

    // 1) projections (WMMA bf16, bias fused)
    proj_gemm<<<dim3((CQK / 16) * (N / 16), B), 32, 0, stream>>>(Wq, bq, sk, qws, CQK, N, C);
    proj_gemm<<<dim3((CQK / 16) * (N / 16), B), 32, 0, stream>>>(Wk, bk, ex, kws, CQK, N, C);
    proj_gemm<<<dim3((C / 16) * (N / 16), B), 32, 0, stream>>>(Wv, bv, ex, vws, C, N, C);

    // 2) energy = scale * q^T k  (single WMMA per 16x16 tile, K=32)
    energy_kernel<<<dim3(N / 16, N / (16 * NQT), B), 32, 0, stream>>>(qws, kws, attn, N, SCALE);

    // 3) softmax rows in place -> final attention output
    softmax_rows<<<B * N, 256, 0, stream>>>(attn, N);

    // 4) texture = v * attn^T (68.7 GF WMMA GEMM), bf16 LDS-staged attention,
    //    fused residual add
    texture_kernel<<<dim3(N / TM, C / 128, B), 256, 0, stream>>>(vws, attn, sk, tex, out, N, C);
}
